// AttentionBlock_88983132438589
// MI455X (gfx1250) — matrix operand order repaired and checked by reading, not compiled, not run on hardware
//
#include <hip/hip_runtime.h>
#include <hip/hip_bf16.h>
#include <math.h>

typedef __attribute__((ext_vector_type(16))) _Float16 v16h;
typedef __attribute__((ext_vector_type(8)))  float    v8f;

// Problem constants
#define Bb   8
#define Ss   1024
#define Hh   1024
#define NHh  16
#define HDd  64

static constexpr int LDST = 40;   // halfs per LDS row (32 data + 8 pad -> 80B stride, 16B aligned)

// ---------------------------------------------------------------------------
// Elementwise prep kernels
// ---------------------------------------------------------------------------
__global__ __launch_bounds__(256) void cvt_f32_to_f16_k(const float* __restrict__ src,
                                                        _Float16* __restrict__ dst, int n) {
  int i = blockIdx.x * 256 + threadIdx.x;
  if (i < n) dst[i] = (_Float16)src[i];
}

__global__ __launch_bounds__(256) void perturb_f32_to_f16_k(const float* __restrict__ src,
                                                            const float* __restrict__ g,
                                                            _Float16* __restrict__ dst, int n) {
  int i = blockIdx.x * 256 + threadIdx.x;
  if (i < n) {
    float w = src[i];
    dst[i] = (_Float16)(w + (*g) * fmaxf(w, 0.0f));
  }
}

__global__ __launch_bounds__(256) void perturb_f32_k(const float* __restrict__ src,
                                                     const float* __restrict__ g,
                                                     float* __restrict__ dst, int n) {
  int i = blockIdx.x * 256 + threadIdx.x;
  if (i < n) {
    float w = src[i];
    dst[i] = w + (*g) * fmaxf(w, 0.0f);
  }
}

// v16[b,s,h*64+d] -> vT[((b*16+h)*64+d)*1024 + s]
__global__ __launch_bounds__(256) void transpose_v_k(const _Float16* __restrict__ v,
                                                     _Float16* __restrict__ vT) {
  long o = (long)blockIdx.x * 256 + threadIdx.x;   // 8*16*64*1024 total
  int  s = (int)(o & 1023);
  int  d = (int)((o >> 10) & 63);
  int  h = (int)((o >> 16) & 15);
  int  b = (int)(o >> 20);
  vT[o] = v[(((long)b * Ss + s) * Hh) + h * HDd + d];
}

// ---------------------------------------------------------------------------
// Generic batched WMMA GEMM:  C[m,n] = sum_k A[m,k] * B[n,k]  (+ bias[n])
//   A: f16 (or f32 converted while staging), row-major, leading dim lda
//   B: f16 stored [N][K] row-major (i.e. column n of the B-matrix = row n)
//   Batch z: base offsets = (z/zi)*S0 + (z%zi)*S1  (element units of each type)
// ---------------------------------------------------------------------------
template <int BM, int BN, int NWM, int NWN, bool AF32, bool OF16, bool BIAS>
__global__ __launch_bounds__(256) void gemm_wmma_k(
    const void* __restrict__ Ap, const _Float16* __restrict__ Bp,
    void* __restrict__ Cp, const float* __restrict__ bias,
    int lda, int ldb, int ldc, int K, int zi,
    long aS0, long aS1, long bS0, long bS1, long cS0, long cS1) {

  constexpr int WTM = BM / (16 * NWM);
  constexpr int WTN = BN / (16 * NWN);

  __shared__ alignas(16) _Float16 As[BM * LDST];
  __shared__ alignas(16) _Float16 Bs[BN * LDST];

  const int tid  = threadIdx.x;
  const int lane = tid & 31;
  const int wave = tid >> 5;
  const int wm   = wave / NWN;
  const int wn   = wave % NWN;
  const int lm   = lane & 15;   // m (A) / n (B) / n (C) within tile
  const int hk   = lane >> 4;   // K-chunk / C row-half selector

  const int  z    = blockIdx.z;
  const long aOff = (long)(z / zi) * aS0 + (long)(z % zi) * aS1;
  const long bOff = (long)(z / zi) * bS0 + (long)(z % zi) * bS1;
  const long cOff = (long)(z / zi) * cS0 + (long)(z % zi) * cS1;

  const int m0 = blockIdx.y * BM;
  const int n0 = blockIdx.x * BN;

  const _Float16* A16 = (const _Float16*)Ap + aOff;
  const float*    A32 = (const float*)Ap + aOff;
  const _Float16* Bm  = Bp + bOff;

  v8f acc[WTM][WTN];
#pragma unroll
  for (int tm = 0; tm < WTM; ++tm)
#pragma unroll
    for (int tn = 0; tn < WTN; ++tn)
      acc[tm][tn] = (v8f){0.f, 0.f, 0.f, 0.f, 0.f, 0.f, 0.f, 0.f};

  union Frag { uint4 u[2]; v16h h; };

  for (int k0 = 0; k0 < K; k0 += 32) {
    // ---- stage A tile (BM x 32 halfs) ----
    constexpr int ITA = (BM * 32) / (256 * 8);
#pragma unroll
    for (int i = 0; i < ITA; ++i) {
      int slot = i * 256 + tid;          // 8 halfs per slot, 4 slots per row
      int r = slot >> 2;
      int c = (slot & 3) << 3;
      if (!AF32) {
        *(uint4*)&As[r * LDST + c] =
            *(const uint4*)&A16[(long)(m0 + r) * lda + k0 + c];
      } else {
        const float* s = &A32[(long)(m0 + r) * lda + k0 + c];
        float4 f0 = *(const float4*)s;
        float4 f1 = *(const float4*)(s + 4);
        union { _Float16 h[8]; uint4 u; } p;
        p.h[0] = (_Float16)f0.x; p.h[1] = (_Float16)f0.y;
        p.h[2] = (_Float16)f0.z; p.h[3] = (_Float16)f0.w;
        p.h[4] = (_Float16)f1.x; p.h[5] = (_Float16)f1.y;
        p.h[6] = (_Float16)f1.z; p.h[7] = (_Float16)f1.w;
        *(uint4*)&As[r * LDST + c] = p.u;
      }
    }
    // ---- stage B tile (BN x 32 halfs) ----
    constexpr int ITB = (BN * 32) / (256 * 8);
#pragma unroll
    for (int i = 0; i < ITB; ++i) {
      int slot = i * 256 + tid;
      int r = slot >> 2;
      int c = (slot & 3) << 3;
      *(uint4*)&Bs[r * LDST + c] =
          *(const uint4*)&Bm[(long)(n0 + r) * ldb + k0 + c];
    }
    __syncthreads();

    // ---- load fragments (ISA §7.12.2 layouts) ----
    Frag af[WTM], bf[WTN];
#pragma unroll
    for (int tm = 0; tm < WTM; ++tm) {
      int row = wm * WTM * 16 + tm * 16 + lm;            // A row m
      const _Float16* base = &As[row * LDST + hk * 8];   // K = 8*hk .. +7
      af[tm].u[0] = *(const uint4*)base;
      af[tm].u[1] = *(const uint4*)(base + 16);          // K = 16+8*hk .. +7
    }
#pragma unroll
    for (int tn = 0; tn < WTN; ++tn) {
      int row = wn * WTN * 16 + tn * 16 + lm;            // B column n
      const _Float16* base = &Bs[row * LDST + hk * 8];   // K = 8*hk .. +7
      bf[tn].u[0] = *(const uint4*)base;
      bf[tn].u[1] = *(const uint4*)(base + 16);          // K = 16+8*hk .. +7
    }

#pragma unroll
    for (int tm = 0; tm < WTM; ++tm)
#pragma unroll
      for (int tn = 0; tn < WTN; ++tn)
        acc[tm][tn] = __builtin_amdgcn_wmma_f32_16x16x32_f16(
            false, af[tm].h, false, bf[tn].h, (short)0, acc[tm][tn], false, false);

    __syncthreads();
  }

  // ---- epilogue: C/D f32 layout -> global ----
#pragma unroll
  for (int tm = 0; tm < WTM; ++tm) {
    int mt = m0 + wm * WTM * 16 + tm * 16 + hk * 8;
#pragma unroll
    for (int tn = 0; tn < WTN; ++tn) {
      int n = n0 + wn * WTN * 16 + tn * 16 + lm;
      float badd = 0.0f;
      if (BIAS) badd = bias[n];
#pragma unroll
      for (int i = 0; i < 8; ++i) {
        float val = acc[tm][tn][i] + badd;
        long idx = cOff + (long)(mt + i) * ldc + n;
        if (OF16) ((_Float16*)Cp)[idx] = (_Float16)val;
        else      ((float*)Cp)[idx]    = val;
      }
    }
  }
}

// ---------------------------------------------------------------------------
// Row softmax in place (rows of 1024 f32), one block per row
// ---------------------------------------------------------------------------
__global__ __launch_bounds__(256) void softmax_rows_k(float* __restrict__ p) {
  const long row = blockIdx.x;
  float4* r = (float4*)(p + row * 1024);
  const int t = threadIdx.x;
  float4 v = r[t];
  __shared__ float red[256];

  float mx = fmaxf(fmaxf(v.x, v.y), fmaxf(v.z, v.w));
  red[t] = mx;
  __syncthreads();
  for (int s = 128; s > 0; s >>= 1) {
    if (t < s) red[t] = fmaxf(red[t], red[t + s]);
    __syncthreads();
  }
  mx = red[0];
  __syncthreads();

  v.x = __expf(v.x - mx); v.y = __expf(v.y - mx);
  v.z = __expf(v.z - mx); v.w = __expf(v.w - mx);
  red[t] = v.x + v.y + v.z + v.w;
  __syncthreads();
  for (int s = 128; s > 0; s >>= 1) {
    if (t < s) red[t] += red[t + s];
    __syncthreads();
  }
  float inv = 1.0f / red[0];
  v.x *= inv; v.y *= inv; v.z *= inv; v.w *= inv;
  r[t] = v;
}

// ---------------------------------------------------------------------------
// Residual + no-affine LayerNorm (eps = 1e-12), one block per row of 1024
// ---------------------------------------------------------------------------
__global__ __launch_bounds__(256) void ln_residual_k(const float* __restrict__ h,
                                                     const float* __restrict__ x,
                                                     float* __restrict__ out) {
  const long row = blockIdx.x;
  const float4* hr = (const float4*)(h + row * 1024);
  const float4* xr = (const float4*)(x + row * 1024);
  float4* orow = (float4*)(out + row * 1024);
  const int t = threadIdx.x;

  float4 a = hr[t], b = xr[t];
  float4 s;
  s.x = a.x + b.x; s.y = a.y + b.y; s.z = a.z + b.z; s.w = a.w + b.w;

  __shared__ float r1[256], r2[256];
  r1[t] = s.x + s.y + s.z + s.w;
  r2[t] = s.x * s.x + s.y * s.y + s.z * s.z + s.w * s.w;
  __syncthreads();
  for (int o = 128; o > 0; o >>= 1) {
    if (t < o) { r1[t] += r1[t + o]; r2[t] += r2[t + o]; }
    __syncthreads();
  }
  float mean = r1[0] * (1.0f / 1024.0f);
  float var  = fmaxf(r2[0] * (1.0f / 1024.0f) - mean * mean, 0.0f);
  float inv  = rsqrtf(var + 1e-12f);
  s.x = (s.x - mean) * inv; s.y = (s.y - mean) * inv;
  s.z = (s.z - mean) * inv; s.w = (s.w - mean) * inv;
  orow[t] = s;
}

// ---------------------------------------------------------------------------
// Host-side launcher
// ---------------------------------------------------------------------------
extern "C" void kernel_launch(void* const* d_in, const int* in_sizes, int n_in,
                              void* d_out, int out_size, void* d_ws, size_t ws_size,
                              hipStream_t stream) {
  (void)in_sizes; (void)n_in; (void)out_size; (void)ws_size;

  const float* hs  = (const float*)d_in[0];
  const float* Wq  = (const float*)d_in[1];
  const float* bq  = (const float*)d_in[2];
  const float* Wk  = (const float*)d_in[3];
  const float* bk  = (const float*)d_in[4];
  const float* Wv  = (const float*)d_in[5];
  const float* bv  = (const float*)d_in[6];
  const float* Wo  = (const float*)d_in[7];
  const float* bo  = (const float*)d_in[8];
  const float* gLN = (const float*)d_in[10];

  const long MR  = (long)Bb * Ss;          // 8192 rows
  const long NHS = (long)MR * Hh;          // 8,388,608 elements [B,S,H]
  const long NW  = (long)Hh * Hh;          // 1,048,576 weight elements

  char* ws = (char*)d_ws;
  // Workspace layout (bytes). hs16 region is later reused for ctx16;
  // q16+k16 region is later reused for f32 h.
  _Float16* hs16  = (_Float16*)(ws + 0);                       // 16 MB
  _Float16* q16   = (_Float16*)(ws + (16l << 20));             // 16 MB
  _Float16* k16   = (_Float16*)(ws + (32l << 20));             // 16 MB
  _Float16* v16   = (_Float16*)(ws + (48l << 20));             // 16 MB
  _Float16* vT16  = (_Float16*)(ws + (64l << 20));             // 16 MB
  _Float16* Wq16  = (_Float16*)(ws + (80l << 20));             // 2 MB
  _Float16* Wk16  = (_Float16*)(ws + (82l << 20));
  _Float16* Wv16  = (_Float16*)(ws + (84l << 20));
  _Float16* Wop16 = (_Float16*)(ws + (86l << 20));
  float*    bo_p  = (float*)   (ws + (88l << 20));             // 4 KB
  _Float16* ctx16 = hs16;                                      // alias (hs16 dead)
  float*    hbuf  = (float*)q16;                               // alias (q16+k16 dead), 32 MB

  float* out_p   = (float*)d_out;                              // [8,1024,1024]
  float* probs_p = (float*)d_out + NHS;                        // [8,16,1024,1024]

  // 1) precision conversion + perturbed output weights
  cvt_f32_to_f16_k<<<dim3((unsigned)(NHS / 256)), 256, 0, stream>>>(hs, hs16, (int)NHS);
  cvt_f32_to_f16_k<<<dim3((unsigned)(NW / 256)), 256, 0, stream>>>(Wq, Wq16, (int)NW);
  cvt_f32_to_f16_k<<<dim3((unsigned)(NW / 256)), 256, 0, stream>>>(Wk, Wk16, (int)NW);
  cvt_f32_to_f16_k<<<dim3((unsigned)(NW / 256)), 256, 0, stream>>>(Wv, Wv16, (int)NW);
  perturb_f32_to_f16_k<<<dim3((unsigned)(NW / 256)), 256, 0, stream>>>(Wo, gLN, Wop16, (int)NW);
  perturb_f32_k<<<dim3(4), 256, 0, stream>>>(bo, gLN, bo_p, Hh);

  // 2) Q/K/V projections: [8192,1024] x [1024,1024]^T + bias -> f16
  {
    dim3 grid(Hh / 128, (unsigned)(MR / 128), 1);
    gemm_wmma_k<128, 128, 4, 2, false, true, true><<<grid, 256, 0, stream>>>(
        hs16, Wq16, q16, bq, Hh, Hh, Hh, Hh, 1, 0, 0, 0, 0, 0, 0);
    gemm_wmma_k<128, 128, 4, 2, false, true, true><<<grid, 256, 0, stream>>>(
        hs16, Wk16, k16, bk, Hh, Hh, Hh, Hh, 1, 0, 0, 0, 0, 0, 0);
    gemm_wmma_k<128, 128, 4, 2, false, true, true><<<grid, 256, 0, stream>>>(
        hs16, Wv16, v16, bv, Hh, Hh, Hh, Hh, 1, 0, 0, 0, 0, 0, 0);
  }

  // 3) V -> per-head transposed layout vT[b,h,d,s]
  transpose_v_k<<<dim3((unsigned)(NHS / 256)), 256, 0, stream>>>(v16, vT16);

  // 4) scores[b,h] = q[b,h] @ k[b,h]^T  (K = 64), f32 straight into probs region
  {
    dim3 grid(Ss / 128, Ss / 128, Bb * NHh);
    gemm_wmma_k<128, 128, 4, 2, false, false, false><<<grid, 256, 0, stream>>>(
        q16, k16, probs_p, nullptr, Hh, Hh, Ss, HDd, NHh,
        (long)Ss * Hh, (long)HDd,            // A(q): b-stride, h-stride (halfs)
        (long)Ss * Hh, (long)HDd,            // B(k)
        (long)NHh * Ss * Ss, (long)Ss * Ss); // C(scores, floats)
  }

  // 5) softmax over each of the 8*16*1024 rows, in place
  softmax_rows_k<<<dim3((unsigned)(Bb * NHh * Ss)), 256, 0, stream>>>(probs_p);

  // 6) ctx[b,h] = probs[b,h] (f32->f16 in LDS) @ vT[b,h]^T : M=1024, N=64, K=1024
  {
    dim3 grid(HDd / 64, Ss / 128, Bb * NHh);
    gemm_wmma_k<128, 64, 8, 1, true, true, false><<<grid, 256, 0, stream>>>(
        probs_p, vT16, ctx16, nullptr, Ss, Ss, Hh, Ss, NHh,
        (long)NHh * Ss * Ss, (long)Ss * Ss,  // A(probs, floats)
        (long)NHh * HDd * Ss, (long)HDd * Ss,// B(vT, halfs)
        (long)Ss * Hh, (long)HDd);           // C(ctx16, halfs): b*S*H + h*64
  }

  // 7) h = ctx @ Wo_p^T + bo_p  -> f32
  {
    dim3 grid(Hh / 128, (unsigned)(MR / 128), 1);
    gemm_wmma_k<128, 128, 4, 2, false, false, true><<<grid, 256, 0, stream>>>(
        ctx16, Wop16, hbuf, bo_p, Hh, Hh, Hh, Hh, 1, 0, 0, 0, 0, 0, 0);
  }

  // 8) out = LayerNorm_nowb(h + hidden_states)
  ln_residual_k<<<dim3((unsigned)MR), 256, 0, stream>>>(hbuf, hs, out_p);
}
